// GNNRateMatrixPredictor_88940182765949
// MI455X (gfx1250) — compile-verified, matrix operand order checked by reading, not run on hardware
//
#include <hip/hip_runtime.h>
#include <hip/hip_bf16.h>
#include <cstddef>

#define BB 16
#define NN 2048
#define EE 65536
#define LL 4

typedef __attribute__((ext_vector_type(16))) _Float16 v16h;
typedef __attribute__((ext_vector_type(8)))  float    v8f;

union H16 { v16h v; float4 q[2]; _Float16 h[16]; };

__device__ __forceinline__ v8f wmma_f16(const H16& a, const H16& b, v8f c) {
  // D = A(16x32 f16) * B(32x16 f16) + C(16x16 f32)
  return __builtin_amdgcn_wmma_f32_16x16x32_f16(false, a.v, false, b.v,
                                                (short)0, c, false, false);
}

// ---------------------------------------------------------------------------
// Pack a (K x 64) f32 weight matrix into WMMA B-fragment layout:
// out[((kt*4+nt)*32 + lane)*16 + i] = W[k, nt*16 + (lane&15)]
// with k = kt*32 + (i>>3)*16 + 8*(lane>>4) + (i&7).
// mode 0: identity rows (K = nkt*32)
// mode 1: layer-0 msg w1 (orig 4 rows -> padded rows 0,1,64,65)
// mode 2: layer-0 upd w1 (orig 66 rows -> padded rows 0,1 and 64..127)
// ---------------------------------------------------------------------------
__global__ void pack_b_kernel(const float* __restrict__ W, _Float16* __restrict__ out,
                              int nkt, int mode) {
  int idx = blockIdx.x * 256 + threadIdx.x;
  int total = nkt * 4 * 32 * 16;
  if (idx >= total) return;
  int i    = idx & 15;
  int lane = (idx >> 4) & 31;
  int nt   = (idx >> 9) & 3;
  int kt   = idx >> 11;
  int k = kt * 32 + ((i >> 3) * 16) + 8 * (lane >> 4) + (i & 7);
  int n = nt * 16 + (lane & 15);
  float v = 0.f;
  if (mode == 0) {
    v = W[k * 64 + n];
  } else if (mode == 1) {
    int ks = -1;
    if (k == 0) ks = 0; else if (k == 1) ks = 1;
    else if (k == 64) ks = 2; else if (k == 65) ks = 3;
    if (ks >= 0) v = W[ks * 64 + n];
  } else {
    int ks = -1;
    if (k < 2) ks = k; else if (k >= 64) ks = k - 62;   // rows 64..127 -> 2..65
    if (ks >= 0) v = W[ks * 64 + n];
  }
  out[idx] = (_Float16)v;
}

// h0[b,n,:] = [mu, t, 0...0]  (f16, width 64)
__global__ void init_h_kernel(const float* __restrict__ mu, const float* __restrict__ t,
                              _Float16* __restrict__ h) {
  int tid = blockIdx.x * 256 + threadIdx.x;
  if (tid >= BB * NN) return;
  int b = tid >> 11;
  _Float16* p = h + (size_t)tid * 64;
  p[0] = (_Float16)mu[tid];
  p[1] = (_Float16)t[b];
  #pragma unroll
  for (int i = 2; i < 64; ++i) p[i] = (_Float16)0.f;
}

// ---------------------------------------------------------------------------
// Message MLP: per wave, 16 edges. X = [h[src] | h[dst]] (16x128 f16)
// GEMM1 (128->64) + b1 + SiLU -> LDS -> GEMM2 (64->64) + b2 -> atomic scatter.
// ---------------------------------------------------------------------------
__global__ __launch_bounds__(128) void msg_kernel(
    const _Float16* __restrict__ h,
    const int* __restrict__ srcA, const int* __restrict__ dstA,
    const _Float16* __restrict__ w1p, const _Float16* __restrict__ w2p,
    const float* __restrict__ b1, const float* __restrict__ b2,
    float* __restrict__ agg) {
  __shared__ _Float16 ldsY[4][16 * 64];
  int wave = threadIdx.x >> 5, lane = threadIdx.x & 31;
  int gt = blockIdx.x * 4 + wave;        // B*E/16 tiles
  int b  = gt >> 12;                     // E/16 = 4096 tiles per batch
  int ebase = (gt & 4095) * 16;
  int m = lane & 15, s = lane >> 4;
  int e = ebase + m;
  int sn = srcA[e], dn = dstA[e];
  const _Float16* hs = h + ((size_t)b * NN + sn) * 64;
  const _Float16* hd = h + ((size_t)b * NN + dn) * 64;

  H16 a[4];
  #pragma unroll
  for (int kt = 0; kt < 4; ++kt) {
    const _Float16* hp = (kt < 2) ? hs : hd;
    int kb = (kt & 1) * 32 + 8 * s;
    a[kt].q[0] = *(const float4*)(hp + kb);
    a[kt].q[1] = *(const float4*)(hp + kb + 16);
  }
  #pragma unroll
  for (int nt = 0; nt < 4; ++nt) {
    v8f c = {};
    #pragma unroll
    for (int kt = 0; kt < 4; ++kt) {
      H16 bf;
      const _Float16* wp = w1p + (((kt * 4 + nt) * 32) + lane) * 16;
      bf.q[0] = *(const float4*)wp;
      bf.q[1] = *(const float4*)(wp + 8);
      c = wmma_f16(a[kt], bf, c);
    }
    float bias = b1[nt * 16 + m];
    #pragma unroll
    for (int r = 0; r < 8; ++r) {
      float x = c[r] + bias;
      ldsY[wave][(r + 8 * s) * 64 + nt * 16 + m] = (_Float16)(x / (1.f + __expf(-x)));
    }
  }
  __syncthreads();
  H16 a2[2];
  #pragma unroll
  for (int kt = 0; kt < 2; ++kt) {
    int kb = kt * 32 + 8 * s;
    a2[kt].q[0] = *(const float4*)(&ldsY[wave][m * 64 + kb]);
    a2[kt].q[1] = *(const float4*)(&ldsY[wave][m * 64 + kb + 16]);
  }
  int drow[8];
  #pragma unroll
  for (int r = 0; r < 8; ++r) drow[r] = dstA[ebase + r + 8 * s];
  #pragma unroll
  for (int nt = 0; nt < 4; ++nt) {
    v8f c = {};
    #pragma unroll
    for (int kt = 0; kt < 2; ++kt) {
      H16 bf;
      const _Float16* wp = w2p + (((kt * 4 + nt) * 32) + lane) * 16;
      bf.q[0] = *(const float4*)wp;
      bf.q[1] = *(const float4*)(wp + 8);
      c = wmma_f16(a2[kt], bf, c);
    }
    float bias = b2[nt * 16 + m];
    #pragma unroll
    for (int r = 0; r < 8; ++r)
      atomicAdd(&agg[((size_t)b * NN + drow[r]) * 64 + nt * 16 + m], c[r] + bias);
  }
}

// ---------------------------------------------------------------------------
// Update MLP: per wave, 16 nodes. X = [h | agg] -> (128->64 SiLU ->64) -> h_out
// ---------------------------------------------------------------------------
__global__ __launch_bounds__(128) void upd_kernel(
    const _Float16* __restrict__ h, const float* __restrict__ agg,
    const _Float16* __restrict__ w1p, const _Float16* __restrict__ w2p,
    const float* __restrict__ b1, const float* __restrict__ b2,
    _Float16* __restrict__ hout) {
  __shared__ _Float16 ldsY[4][16 * 64];
  int wave = threadIdx.x >> 5, lane = threadIdx.x & 31;
  int gt = blockIdx.x * 4 + wave;        // B*N/16 tiles
  int b  = gt >> 7;                      // N/16 = 128 tiles per batch
  int nbase = (gt & 127) * 16;
  int m = lane & 15, s = lane >> 4;
  int node = nbase + m;
  const _Float16* hp = h   + ((size_t)b * NN + node) * 64;
  const float*    ap = agg + ((size_t)b * NN + node) * 64;

  H16 a[4];
  #pragma unroll
  for (int kt = 0; kt < 2; ++kt) {
    int kb = kt * 32 + 8 * s;
    a[kt].q[0] = *(const float4*)(hp + kb);
    a[kt].q[1] = *(const float4*)(hp + kb + 16);
  }
  #pragma unroll
  for (int kt = 2; kt < 4; ++kt) {
    int kb = (kt - 2) * 32 + 8 * s;
    #pragma unroll
    for (int j = 0; j < 8; ++j) a[kt].h[j]     = (_Float16)ap[kb + j];
    #pragma unroll
    for (int j = 0; j < 8; ++j) a[kt].h[8 + j] = (_Float16)ap[kb + 16 + j];
  }
  #pragma unroll
  for (int nt = 0; nt < 4; ++nt) {
    v8f c = {};
    #pragma unroll
    for (int kt = 0; kt < 4; ++kt) {
      H16 bf;
      const _Float16* wp = w1p + (((kt * 4 + nt) * 32) + lane) * 16;
      bf.q[0] = *(const float4*)wp;
      bf.q[1] = *(const float4*)(wp + 8);
      c = wmma_f16(a[kt], bf, c);
    }
    float bias = b1[nt * 16 + m];
    #pragma unroll
    for (int r = 0; r < 8; ++r) {
      float x = c[r] + bias;
      ldsY[wave][(r + 8 * s) * 64 + nt * 16 + m] = (_Float16)(x / (1.f + __expf(-x)));
    }
  }
  __syncthreads();
  H16 a2[2];
  #pragma unroll
  for (int kt = 0; kt < 2; ++kt) {
    int kb = kt * 32 + 8 * s;
    a2[kt].q[0] = *(const float4*)(&ldsY[wave][m * 64 + kb]);
    a2[kt].q[1] = *(const float4*)(&ldsY[wave][m * 64 + kb + 16]);
  }
  #pragma unroll
  for (int nt = 0; nt < 4; ++nt) {
    v8f c = {};
    #pragma unroll
    for (int kt = 0; kt < 2; ++kt) {
      H16 bf;
      const _Float16* wp = w2p + (((kt * 4 + nt) * 32) + lane) * 16;
      bf.q[0] = *(const float4*)wp;
      bf.q[1] = *(const float4*)(wp + 8);
      c = wmma_f16(a2[kt], bf, c);
    }
    float bias = b2[nt * 16 + m];
    #pragma unroll
    for (int r = 0; r < 8; ++r)
      hout[((size_t)b * NN + nbase + r + 8 * s) * 64 + nt * 16 + m] = (_Float16)(c[r] + bias);
  }
}

// ---------------------------------------------------------------------------
// Edge readout: GEMM1 (128->64) + SiLU (WMMA), 64->1 dot (VALU), softplus,
// scatter into rate matrix + row-sum atomics.
// ---------------------------------------------------------------------------
__global__ __launch_bounds__(128) void edge_kernel(
    const _Float16* __restrict__ h,
    const int* __restrict__ srcA, const int* __restrict__ dstA,
    const _Float16* __restrict__ w1p, const float* __restrict__ b1,
    const float* __restrict__ w2, const float* __restrict__ b2,
    float* __restrict__ rm, float* __restrict__ rowsum) {
  __shared__ float ldsY[4][16 * 64];
  int wave = threadIdx.x >> 5, lane = threadIdx.x & 31;
  int gt = blockIdx.x * 4 + wave;
  int b  = gt >> 12;
  int ebase = (gt & 4095) * 16;
  int m = lane & 15, s = lane >> 4;
  int e = ebase + m;
  int sn = srcA[e], dn = dstA[e];
  const _Float16* hs = h + ((size_t)b * NN + sn) * 64;
  const _Float16* hd = h + ((size_t)b * NN + dn) * 64;

  H16 a[4];
  #pragma unroll
  for (int kt = 0; kt < 4; ++kt) {
    const _Float16* hp = (kt < 2) ? hs : hd;
    int kb = (kt & 1) * 32 + 8 * s;
    a[kt].q[0] = *(const float4*)(hp + kb);
    a[kt].q[1] = *(const float4*)(hp + kb + 16);
  }
  #pragma unroll
  for (int nt = 0; nt < 4; ++nt) {
    v8f c = {};
    #pragma unroll
    for (int kt = 0; kt < 4; ++kt) {
      H16 bf;
      const _Float16* wp = w1p + (((kt * 4 + nt) * 32) + lane) * 16;
      bf.q[0] = *(const float4*)wp;
      bf.q[1] = *(const float4*)(wp + 8);
      c = wmma_f16(a[kt], bf, c);
    }
    float bias = b1[nt * 16 + m];
    #pragma unroll
    for (int r = 0; r < 8; ++r) {
      float x = c[r] + bias;
      ldsY[wave][(r + 8 * s) * 64 + nt * 16 + m] = x / (1.f + __expf(-x));
    }
  }
  __syncthreads();
  if (lane < 16) {
    int ee = ebase + lane;
    int s2 = srcA[ee], d2 = dstA[ee];
    float acc = b2[0];
    const float* yr = &ldsY[wave][lane * 64];
    #pragma unroll 4
    for (int n = 0; n < 64; ++n) acc += yr[n] * w2[n];
    float rate = (acc > 20.f) ? acc : log1pf(__expf(acc));
    rm[((size_t)b * NN + s2) * NN + d2] = rate;
    atomicAdd(&rowsum[b * NN + s2], rate);
  }
}

__global__ void diag_kernel(const float* __restrict__ rowsum, float* __restrict__ rm) {
  int tid = blockIdx.x * 256 + threadIdx.x;
  if (tid >= BB * NN) return;
  int b = tid >> 11, i = tid & 2047;
  rm[((size_t)b * NN + i) * NN + i] = -rowsum[tid];
}

// ---------------------------------------------------------------------------
// Host side
// ---------------------------------------------------------------------------
static void resolve_mlp(void* const* din, const int* sizes, int base,
                        long fi, long fh, long fo,
                        const float** w1, const float** b1,
                        const float** w2, const float** b2) {
  const float* p[4]; long s[4]; bool used[4] = {false, false, false, false};
  for (int j = 0; j < 4; ++j) { p[j] = (const float*)din[base + j]; s[j] = sizes[base + j]; }
  auto take = [&](long sz) -> const float* {
    for (int j = 0; j < 4; ++j)
      if (!used[j] && s[j] == sz) { used[j] = true; return p[j]; }
    return nullptr;
  };
  *w1 = take(fi * fh);
  *b2 = nullptr; *w2 = nullptr;
  if (fo != fh)      *b2 = take(fo);       // unique-size b2 (edge MLP)
  if (fh * fo != fh) *w2 = take(fh * fo);  // unique-size w2 (square MLPs)
  const float* rem[4]; int nr = 0;
  for (int j = 0; j < 4; ++j) if (!used[j]) rem[nr++] = p[j];
  int k = 0;
  *b1 = rem[k++];
  if (!*w2) *w2 = rem[k++];
  if (!*b2) *b2 = rem[k++];
}

extern "C" void kernel_launch(void* const* d_in, const int* in_sizes, int n_in,
                              void* d_out, int out_size, void* d_ws, size_t ws_size,
                              hipStream_t stream) {
  (void)n_in; (void)ws_size;
  const float* mu = (const float*)d_in[0];
  const float* t  = (const float*)d_in[1];
  const int* src  = (const int*)d_in[2];
  const int* dst  = src + EE;

  const float *mw1[LL], *mb1[LL], *mw2[LL], *mb2[LL];
  const float *uw1[LL], *ub1[LL], *uw2[LL], *ub2[LL];
  for (int l = 0; l < LL; ++l) {
    long fi = (l == 0) ? 2 : 64;
    resolve_mlp(d_in, in_sizes, 3 + 8 * l,     2 * fi,  64, 64, &mw1[l], &mb1[l], &mw2[l], &mb2[l]);
    resolve_mlp(d_in, in_sizes, 3 + 8 * l + 4, fi + 64, 64, 64, &uw1[l], &ub1[l], &uw2[l], &ub2[l]);
  }
  const float *ew1, *eb1, *ew2, *eb2;
  resolve_mlp(d_in, in_sizes, 3 + 8 * LL, 128, 64, 1, &ew1, &eb1, &ew2, &eb2);

  char* ws = (char*)d_ws;
  size_t hBytes = (size_t)BB * NN * 64 * sizeof(_Float16);              // 4 MB
  _Float16* hA = (_Float16*)ws;
  _Float16* hB = (_Float16*)(ws + hBytes);
  float* agg    = (float*)(ws + 2 * hBytes);                            // 8 MB
  float* rowsum = (float*)(ws + 2 * hBytes + (size_t)BB * NN * 64 * sizeof(float));
  _Float16* pp  = (_Float16*)((char*)rowsum + (size_t)BB * NN * sizeof(float));
  _Float16 *mw1p[LL], *mw2p[LL], *uw1p[LL], *uw2p[LL];
  for (int l = 0; l < LL; ++l) {
    mw1p[l] = pp; pp += 8192;
    mw2p[l] = pp; pp += 4096;
    uw1p[l] = pp; pp += 8192;
    uw2p[l] = pp; pp += 4096;
  }
  _Float16* ew1p = pp;

  for (int l = 0; l < LL; ++l) {
    pack_b_kernel<<<32, 256, 0, stream>>>(mw1[l], mw1p[l], 4, l == 0 ? 1 : 0);
    pack_b_kernel<<<16, 256, 0, stream>>>(mw2[l], mw2p[l], 2, 0);
    pack_b_kernel<<<32, 256, 0, stream>>>(uw1[l], uw1p[l], 4, l == 0 ? 2 : 0);
    pack_b_kernel<<<16, 256, 0, stream>>>(uw2[l], uw2p[l], 2, 0);
  }
  pack_b_kernel<<<32, 256, 0, stream>>>(ew1, ew1p, 4, 0);

  init_h_kernel<<<(BB * NN + 255) / 256, 256, 0, stream>>>(mu, t, hA);

  _Float16* hc = hA; _Float16* hn = hB;
  for (int l = 0; l < LL; ++l) {
    hipMemsetAsync(agg, 0, (size_t)BB * NN * 64 * sizeof(float), stream);
    msg_kernel<<<BB * EE / 16 / 4, 128, 0, stream>>>(hc, src, dst, mw1p[l], mw2p[l],
                                                     mb1[l], mb2[l], agg);
    upd_kernel<<<BB * NN / 16 / 4, 128, 0, stream>>>(hc, agg, uw1p[l], uw2p[l],
                                                     ub1[l], ub2[l], hn);
    _Float16* tmp = hc; hc = hn; hn = tmp;
  }

  hipMemsetAsync(rowsum, 0, (size_t)BB * NN * sizeof(float), stream);
  hipMemsetAsync(d_out, 0, (size_t)out_size * sizeof(float), stream);
  edge_kernel<<<BB * EE / 16 / 4, 128, 0, stream>>>(hc, src, dst, ew1p, eb1, ew2, eb2,
                                                    (float*)d_out, rowsum);
  diag_kernel<<<(BB * NN + 255) / 256, 256, 0, stream>>>(rowsum, (float*)d_out);
}
